// CP_Attention_Skill_Prompt_1563368096229
// MI455X (gfx1250) — matrix operand order repaired and checked by reading, not compiled, not run on hardware
//
#include <hip/hip_runtime.h>
#include <hip/hip_bf16.h>
#include <stdint.h>

// ---------------------------------------------------------------------------
// Shapes (fixed by the reference): B=4, N=2048, D=1024, H=16, hd=64, r=R=64
// ---------------------------------------------------------------------------
#define SEQ   2048
#define DMODEL 1024
#define BATCH 4
#define HEADS 16
#define HDIM  64
#define MROWS (BATCH * SEQ)        // 8192
#define ATT_SCALE 0.125f           // 1/sqrt(64)

typedef __attribute__((ext_vector_type(16))) __bf16 v16bf;
typedef __attribute__((ext_vector_type(8)))  float  v8f;

union FragB16 { v16bf v; uint4 q[2]; unsigned short s[16]; };

__device__ __forceinline__ unsigned short f2bf(float f) {
    unsigned int u = __float_as_uint(f);
    u += 0x7FFFu + ((u >> 16) & 1u);          // round-to-nearest-even
    return (unsigned short)(u >> 16);
}

__device__ __forceinline__ v8f wmma_bf16(const FragB16& a, const FragB16& b, v8f c) {
    return __builtin_amdgcn_wmma_f32_16x16x32_bf16(
        /*neg_a=*/false, a.v, /*neg_b=*/false, b.v,
        /*c_mod=*/(short)0, c, /*reuse_a=*/false, /*reuse_b=*/false);
}

// ---------------------------------------------------------------------------
// 1) f32 -> bf16 cast, 8 elements/thread (vectorized b128 in / b128 out)
// ---------------------------------------------------------------------------
__global__ void cast_f32_bf16_v8(const float* __restrict__ in,
                                 unsigned short* __restrict__ out, int n8) {
    int i = blockIdx.x * blockDim.x + threadIdx.x;
    if (i >= n8) return;
    const float4* p = (const float4*)in + (size_t)i * 2;
    float4 a = p[0], b = p[1];
    union { unsigned short s[8]; uint4 q; } u;
    u.s[0] = f2bf(a.x); u.s[1] = f2bf(a.y); u.s[2] = f2bf(a.z); u.s[3] = f2bf(a.w);
    u.s[4] = f2bf(b.x); u.s[5] = f2bf(b.y); u.s[6] = f2bf(b.z); u.s[7] = f2bf(b.w);
    ((uint4*)out)[i] = u.q;
}

// ---------------------------------------------------------------------------
// 2) CPc = einsum('xyr,rf->xyf', CP_C, CP_attention) split into 4 (64x64) mats
// ---------------------------------------------------------------------------
__global__ void cpc_kernel(const float* __restrict__ CP_C,     // (64,64,64)
                           const float* __restrict__ CP_att,   // (64,4)
                           float* __restrict__ cp) {            // (4,64,64)
    int t = blockIdx.x * blockDim.x + threadIdx.x;              // 4096
    if (t >= 4096) return;
    const float* c = CP_C + (size_t)t * 64;
    float a0 = 0.f, a1 = 0.f, a2 = 0.f, a3 = 0.f;
#pragma unroll 8
    for (int rr = 0; rr < 64; rr++) {
        float cv = c[rr];
        a0 += cv * CP_att[rr * 4 + 0];
        a1 += cv * CP_att[rr * 4 + 1];
        a2 += cv * CP_att[rr * 4 + 2];
        a3 += cv * CP_att[rr * 4 + 3];
    }
    cp[0 * 4096 + t] = a0; cp[1 * 4096 + t] = a1;
    cp[2 * 4096 + t] = a2; cp[3 * 4096 + t] = a3;
}

// ---------------------------------------------------------------------------
// 3) T[a] = CP_U_W (1024x64) @ cp[a] (64x64)   -> T (4,1024,64)
// ---------------------------------------------------------------------------
__global__ void tmat_kernel(const float* __restrict__ U,
                            const float* __restrict__ cp,
                            float* __restrict__ T) {
    int t = blockIdx.x * blockDim.x + threadIdx.x;              // 4*1024*64
    if (t >= 4 * 1024 * 64) return;
    int a = t >> 16; int rem = t & 65535;
    int d = rem >> 6; int j = rem & 63;
    const float* u = U + (size_t)d * 64;
    const float* c = cp + a * 4096 + j;
    float acc = 0.f;
#pragma unroll 8
    for (int i = 0; i < 64; i++) acc += u[i] * c[i * 64];
    T[t] = acc;                                                 // [a][d][j]
}

// ---------------------------------------------------------------------------
// 4) Folded weights, transposed + bf16:
//    Wt[a][n][k] = W_a[k][n] + sum_j T[a][k][j] * CP_V_W[j][n]
// ---------------------------------------------------------------------------
__global__ void wefft_kernel(const float* __restrict__ Wq, const float* __restrict__ Wk,
                             const float* __restrict__ Wv, const float* __restrict__ Wp,
                             const float* __restrict__ T,  const float* __restrict__ Vw,
                             unsigned short* __restrict__ Wt) {
    int t = blockIdx.x * blockDim.x + threadIdx.x;              // 4*1024*1024
    if (t >= 4 * 1024 * 1024) return;
    int a = t >> 20; int rem = t & 1048575;
    int n = rem >> 10; int k = rem & 1023;
    const float* W = (a == 0) ? Wq : (a == 1) ? Wk : (a == 2) ? Wv : Wp;
    const float* tr = T + (size_t)a * 65536 + (size_t)k * 64;
    float acc = W[(size_t)k * DMODEL + n];
#pragma unroll 8
    for (int j = 0; j < 64; j++) acc += tr[j] * Vw[(size_t)j * DMODEL + n];
    Wt[(size_t)a * 1048576 + (size_t)n * DMODEL + k] = f2bf(acc);
}

// ---------------------------------------------------------------------------
// 5) Main GEMM: Out(M x 1024) = A(M x 1024, bf16 row-major) @ Wt^T
//    Wt stored (N x K). 256 threads = 8 waves, WG tile 128 x 256,
//    wave tile 64 x 64 (4x4 WMMA tiles -> 16 wmma / 16 b128 loads per K step),
//    K step 32.
//    mode 0: f32 row-major + bias (final proj -> d_out)
//    mode 1: bf16 (B,H,N,hd)            (Q with scale, K)
//    mode 2: bf16 (B,H,hd,N) transposed (V)
// ---------------------------------------------------------------------------
__global__ __launch_bounds__(256)
void gemm_bf16(const unsigned short* __restrict__ A,
               const unsigned short* __restrict__ Bt,
               void* __restrict__ Out,
               const float* __restrict__ bias,
               float out_scale, int mode) {
    const int lane = threadIdx.x & 31;
    const int wave = threadIdx.x >> 5;
    const int lm = lane & 15, lh = lane >> 4;
    const int wm = wave & 1;          // 2 waves along M
    const int wn = wave >> 1;         // 4 waves along N
    const int mBase = blockIdx.y * 128 + wm * 64;
    const int nBase = blockIdx.x * 256 + wn * 64;

    v8f acc[4][4];
#pragma unroll
    for (int i = 0; i < 4; i++)
#pragma unroll
        for (int j = 0; j < 4; j++) { v8f z = {}; acc[i][j] = z; }

    const unsigned short* ar[4];
    const unsigned short* br[4];
#pragma unroll
    for (int t = 0; t < 4; t++) {
        ar[t] = A  + (size_t)(mBase + t * 16 + lm) * DMODEL;
        br[t] = Bt + (size_t)(nBase + t * 16 + lm) * DMODEL;
    }

    for (int kk = 0; kk < DMODEL; kk += 32) {
        FragB16 bf[4];
#pragma unroll
        for (int tn = 0; tn < 4; tn++) {
            bf[tn].q[0] = *(const uint4*)(br[tn] + kk + 8 * lh);
            bf[tn].q[1] = *(const uint4*)(br[tn] + kk + 16 + 8 * lh);
        }
#pragma unroll
        for (int tm = 0; tm < 4; tm++) {
            FragB16 af;
            af.q[0] = *(const uint4*)(ar[tm] + kk + 8 * lh);
            af.q[1] = *(const uint4*)(ar[tm] + kk + 16 + 8 * lh);
#pragma unroll
            for (int tn = 0; tn < 4; tn++)
                acc[tm][tn] = wmma_bf16(af, bf[tn], acc[tm][tn]);
        }
    }

#pragma unroll
    for (int tm = 0; tm < 4; tm++)
#pragma unroll
        for (int tn = 0; tn < 4; tn++) {
            int n = nBase + tn * 16 + lm;
            int mt = mBase + tm * 16;
            if (mode == 0) {
                float bv = bias ? bias[n] : 0.f;
                float* O = (float*)Out;
#pragma unroll
                for (int v = 0; v < 8; v++) {
                    int m = mt + v + 8 * lh;
                    O[(size_t)m * DMODEL + n] = acc[tm][tn][v] * out_scale + bv;
                }
            } else if (mode == 1) {
                unsigned short* O = (unsigned short*)Out;
                int h = n >> 6, hd = n & 63;
#pragma unroll
                for (int v = 0; v < 8; v++) {
                    int m = mt + v + 8 * lh;
                    int b = m >> 11, q = m & (SEQ - 1);
                    O[(((size_t)(b * HEADS + h) * SEQ + q) << 6) + hd] =
                        f2bf(acc[tm][tn][v] * out_scale);
                }
            } else {  // mode 2: V transposed (B,H,hd,N)
                unsigned short* O = (unsigned short*)Out;
                int h = n >> 6, hd = n & 63;
                int m0 = mt + 8 * lh;
                int b = m0 >> 11, q0 = m0 & (SEQ - 1);
                union { unsigned short s[8]; uint4 q; } pk;
#pragma unroll
                for (int v = 0; v < 8; v++) pk.s[v] = f2bf(acc[tm][tn][v] * out_scale);
                *(uint4*)&O[((size_t)(b * HEADS + h) * HDIM + hd) * SEQ + q0] = pk.q;
            }
        }
}

// ---------------------------------------------------------------------------
// 6) Flash attention: grid = B*H * (SEQ/64), block = 128 (4 waves).
//    Each wave: 16 q-rows, streams keys in chunks of 32, online softmax,
//    P staged through LDS to convert C-layout -> A-fragment layout.
//    Q pre-scaled by ATT_SCALE at GEMM epilogue. Output x in bf16 (B,N,D).
// ---------------------------------------------------------------------------
__global__ __launch_bounds__(128)
void attn_fused(const unsigned short* __restrict__ Qb,
                const unsigned short* __restrict__ Kb,
                const unsigned short* __restrict__ VT,
                unsigned short* __restrict__ X) {
    __shared__ __align__(16) unsigned short ldsP[4][16][48];  // 96B rows (16B aligned)
    const int lane = threadIdx.x & 31;
    const int wave = threadIdx.x >> 5;
    const int lm = lane & 15, lh = lane >> 4;
    const int bh = blockIdx.x >> 5;                 // b*16 + h
    const int q0 = (blockIdx.x & 31) * 64 + wave * 16;

    // persistent Q A-fragments (hd 0..31 and 32..63)
    const unsigned short* Qrow = Qb + ((size_t)bh * SEQ + q0 + lm) * HDIM;
    FragB16 qa0, qa1;
    qa0.q[0] = *(const uint4*)(Qrow + 8 * lh);
    qa0.q[1] = *(const uint4*)(Qrow + 16 + 8 * lh);
    qa1.q[0] = *(const uint4*)(Qrow + 32 + 8 * lh);
    qa1.q[1] = *(const uint4*)(Qrow + 48 + 8 * lh);

    v8f acc[4];
#pragma unroll
    for (int i = 0; i < 4; i++) { v8f z = {}; acc[i] = z; }
    float mrun[8], lrun[8];
#pragma unroll
    for (int v = 0; v < 8; v++) { mrun[v] = -1e30f; lrun[v] = 0.f; }

    for (int kc = 0; kc < SEQ; kc += 32) {
        // ---- S = Q @ K^T for 32 keys (two 16-key C tiles) ----
        v8f s0 = {}, s1 = {};
        {
            const unsigned short* Kr0 = Kb + ((size_t)bh * SEQ + kc + lm) * HDIM;
            const unsigned short* Kr1 = Kr0 + 16 * HDIM;
            FragB16 kb;
            kb.q[0] = *(const uint4*)(Kr0 + 8 * lh);
            kb.q[1] = *(const uint4*)(Kr0 + 16 + 8 * lh);
            s0 = wmma_bf16(qa0, kb, s0);
            kb.q[0] = *(const uint4*)(Kr0 + 32 + 8 * lh);
            kb.q[1] = *(const uint4*)(Kr0 + 48 + 8 * lh);
            s0 = wmma_bf16(qa1, kb, s0);
            kb.q[0] = *(const uint4*)(Kr1 + 8 * lh);
            kb.q[1] = *(const uint4*)(Kr1 + 16 + 8 * lh);
            s1 = wmma_bf16(qa0, kb, s1);
            kb.q[0] = *(const uint4*)(Kr1 + 32 + 8 * lh);
            kb.q[1] = *(const uint4*)(Kr1 + 48 + 8 * lh);
            s1 = wmma_bf16(qa1, kb, s1);
        }
        // ---- online softmax; rows 0-7 in lanes 0-15, rows 8-15 in 16-31 ----
#pragma unroll
        for (int v = 0; v < 8; v++) {
            float mx = fmaxf(s0[v], s1[v]);
            mx = fmaxf(mx, __shfl_xor(mx, 1, 32));
            mx = fmaxf(mx, __shfl_xor(mx, 2, 32));
            mx = fmaxf(mx, __shfl_xor(mx, 4, 32));
            mx = fmaxf(mx, __shfl_xor(mx, 8, 32));
            float mnew = fmaxf(mrun[v], mx);
            float r  = __expf(mrun[v] - mnew);
            float p0 = __expf(s0[v] - mnew);
            float p1 = __expf(s1[v] - mnew);
            float ps = p0 + p1;
            ps += __shfl_xor(ps, 1, 32);
            ps += __shfl_xor(ps, 2, 32);
            ps += __shfl_xor(ps, 4, 32);
            ps += __shfl_xor(ps, 8, 32);
            lrun[v] = lrun[v] * r + ps;
            mrun[v] = mnew;
#pragma unroll
            for (int nt = 0; nt < 4; nt++) acc[nt][v] *= r;
            int row = v + 8 * lh;
            ldsP[wave][row][lm]      = f2bf(p0);
            ldsP[wave][row][16 + lm] = f2bf(p1);
        }
        asm volatile("" ::: "memory");   // keep ds_store before ds_load (LDS in-order per wave)
        FragB16 pa;
        pa.q[0] = *(const uint4*)&ldsP[wave][lm][8 * lh];
        pa.q[1] = *(const uint4*)&ldsP[wave][lm][16 + 8 * lh];
        asm volatile("" ::: "memory");
        // ---- O += P @ V over 32 keys, 4 hd tiles ----
#pragma unroll
        for (int nt = 0; nt < 4; nt++) {
            const unsigned short* Vr =
                VT + ((size_t)bh * HDIM + nt * 16 + lm) * SEQ + kc + 8 * lh;
            FragB16 vb;
            vb.q[0] = *(const uint4*)(Vr);
            vb.q[1] = *(const uint4*)(Vr + 16);
            acc[nt] = wmma_bf16(pa, vb, acc[nt]);
        }
    }

    // ---- normalize + store x (B,N,D) bf16 ----
    int b = bh >> 4, h = bh & 15;
#pragma unroll
    for (int v = 0; v < 8; v++) {
        float inv = 1.f / lrun[v];
        int q = q0 + v + 8 * lh;
        size_t base = ((size_t)b * SEQ + q) * DMODEL + h * HDIM;
#pragma unroll
        for (int nt = 0; nt < 4; nt++)
            X[base + nt * 16 + lm] = f2bf(acc[nt][v] * inv);
    }
}

// ---------------------------------------------------------------------------
// Launch
// ---------------------------------------------------------------------------
extern "C" void kernel_launch(void* const* d_in, const int* in_sizes, int n_in,
                              void* d_out, int out_size, void* d_ws, size_t ws_size,
                              hipStream_t stream) {
    const float* in_q   = (const float*)d_in[0];
    const float* in_k   = (const float*)d_in[1];
    const float* in_v   = (const float*)d_in[2];
    const float* Wq     = (const float*)d_in[3];
    const float* Wk     = (const float*)d_in[4];
    const float* Wv     = (const float*)d_in[5];
    const float* Wproj  = (const float*)d_in[6];
    const float* bproj  = (const float*)d_in[7];
    const float* CP_att = (const float*)d_in[8];
    const float* CP_C   = (const float*)d_in[9];
    const float* CP_U   = (const float*)d_in[10];
    const float* CP_V   = (const float*)d_in[11];

    char* ws = (char*)d_ws;
    size_t off = 0;
    auto alloc = [&](size_t bytes) {
        size_t o = off; off = (off + bytes + 255) & ~(size_t)255; return o;
    };
    const size_t nElem = (size_t)MROWS * DMODEL;                 // 8.39M
    float*          cp  = (float*)(ws + alloc(4 * 4096 * sizeof(float)));
    float*          T   = (float*)(ws + alloc(4 * 1024 * 64 * sizeof(float)));
    unsigned short* Wt  = (unsigned short*)(ws + alloc(4ull * 1024 * 1024 * 2));
    unsigned short* Xq  = (unsigned short*)(ws + alloc(nElem * 2));
    unsigned short* Xk  = (unsigned short*)(ws + alloc(nElem * 2));
    unsigned short* Xv  = (unsigned short*)(ws + alloc(nElem * 2));
    unsigned short* Qb  = (unsigned short*)(ws + alloc(nElem * 2));
    unsigned short* Kbm = (unsigned short*)(ws + alloc(nElem * 2));
    unsigned short* VT  = (unsigned short*)(ws + alloc(nElem * 2));
    unsigned short* Xo  = (unsigned short*)(ws + alloc(nElem * 2));

    // 1) casts
    int n8 = (int)(nElem / 8);
    cast_f32_bf16_v8<<<n8 / 256, 256, 0, stream>>>(in_q, Xq, n8);
    cast_f32_bf16_v8<<<n8 / 256, 256, 0, stream>>>(in_k, Xk, n8);
    cast_f32_bf16_v8<<<n8 / 256, 256, 0, stream>>>(in_v, Xv, n8);

    // 2-4) fold adapters into weights
    cpc_kernel  <<<16,   256, 0, stream>>>(CP_C, CP_att, cp);
    tmat_kernel <<<1024, 256, 0, stream>>>(CP_U, cp, T);
    wefft_kernel<<<16384,256, 0, stream>>>(Wq, Wk, Wv, Wproj, T, CP_V, Wt);

    // 5) QKV GEMMs (Q pre-scaled by 1/sqrt(hd); V written transposed)
    dim3 gg(DMODEL / 256, MROWS / 128);
    gemm_bf16<<<gg, 256, 0, stream>>>(Xq, Wt + 0 * 1048576, Qb,  nullptr, ATT_SCALE, 1);
    gemm_bf16<<<gg, 256, 0, stream>>>(Xk, Wt + 1 * 1048576, Kbm, nullptr, 1.0f, 1);
    gemm_bf16<<<gg, 256, 0, stream>>>(Xv, Wt + 2 * 1048576, VT,  nullptr, 1.0f, 2);

    // 6) fused attention
    attn_fused<<<BATCH * HEADS * (SEQ / 64), 128, 0, stream>>>(Qb, Kbm, VT, Xo);

    // 7) output projection (folded adapter) + bias -> f32 d_out
    gemm_bf16<<<gg, 256, 0, stream>>>(Xo, Wt + 3 * 1048576, d_out, bproj, 1.0f, 0);
}
